// GroupConvAttention_80625126081359
// MI455X (gfx1250) — matrix operand order repaired and checked by reading, compiled, not run on hardware
//
#include <hip/hip_runtime.h>

// CDNA5 / gfx1250, wave32. WMMA f16 path (codegen-confirmed builtin).
typedef __attribute__((ext_vector_type(16))) _Float16 v16h;
typedef __attribute__((ext_vector_type(8)))  float    v8f;
typedef __attribute__((ext_vector_type(4)))  float    v4f;
typedef int v4i __attribute__((vector_size(16)));

#define BB   8
#define CC   16
#define HH   128
#define WW   128
#define HS   64
#define PP   4096          // hs*ws keys / queries
#define NN   256           // C * vk * vk
#define KSTEP 32           // WMMA K per step (f16 16x16x32)
#define KT   (PP / KSTEP)  // 128 K-tiles
#define BTILE_ELEMS (16 * 32 * 16)   // [nb][lane][e] f16 = 8192 elems = 16 KB

#define AS1 __attribute__((address_space(1)))
#define AS3 __attribute__((address_space(3)))

#if __has_builtin(__builtin_amdgcn_global_load_async_to_lds_b128)
#define HAVE_ASYNC_LDS 1
#else
#define HAVE_ASYNC_LDS 0
#endif

// ---------------------------------------------------------------------------
// Kernel 1: extract replication-padded 4x4 patches of x into V, stored
// directly in the WMMA B-fragment swizzled layout:
//   Vw[b][kt][nb][lane][e]  (f16)
//   B matrix 32x16 per (kt, nb):  N = nb*16 + (lane&15)
//                                 K = kt*32 + (lane>>4)*16 + e
// ---------------------------------------------------------------------------
__global__ void build_patches_kernel(const float* __restrict__ x,
                                     _Float16* __restrict__ Vw) {
    long gid = (long)blockIdx.x * blockDim.x + threadIdx.x;  // 8,388,608 total
    int e    = (int)(gid & 15);
    int lane = (int)((gid >> 4) & 31);
    int nb   = (int)((gid >> 9) & 15);
    int kt   = (int)((gid >> 13) & 127);
    int b    = (int)(gid >> 20);

    int k = kt * KSTEP + ((lane >> 4) << 4) + e;   // key / patch index p
    int n = nb * 16 + (lane & 15);                 // output column (c,ky,kx)

    int i  = k >> 6;          // patch row   (0..63)
    int j  = k & 63;          // patch col   (0..63)
    int c  = n >> 4;          // channel     (0..15)
    int ky = (n >> 2) & 3;    // kernel y    (0..3)
    int kx = n & 3;           // kernel x    (0..3)

    int yy = 2 * i + ky - 1;  // replication pad clamp
    yy = yy < 0 ? 0 : (yy > HH - 1 ? HH - 1 : yy);
    int xx = 2 * j + kx - 1;
    xx = xx < 0 ? 0 : (xx > WW - 1 ? WW - 1 : xx);

    float v = x[(((long)b * CC + c) * HH + yy) * WW + xx];
    Vw[gid] = (_Float16)v;   // writes are fully coalesced (gid-linear)
}

// ---------------------------------------------------------------------------
// Stage one 16 KB B tile (1024 x 16B) into LDS with 256 threads via the
// CDNA5 direct global->LDS async path (ASYNCcnt); fallback: register copy.
// ---------------------------------------------------------------------------
__device__ __forceinline__ void stage_btile(const uint4* __restrict__ src,
                                            uint4* dst, int tid) {
#if HAVE_ASYNC_LDS
#pragma unroll
    for (int t = 0; t < 4; ++t) {
        __builtin_amdgcn_global_load_async_to_lds_b128(
            (v4i AS1*)(src + tid + 256 * t),
            (v4i AS3*)(dst + tid + 256 * t),
            /*offset=*/0, /*cpol=*/0);
    }
#else
#pragma unroll
    for (int t = 0; t < 4; ++t) dst[tid + 256 * t] = src[tid + 256 * t];
#endif
}

__device__ __forceinline__ void wait_stage() {
#if HAVE_ASYNC_LDS
#if __has_builtin(__builtin_amdgcn_s_wait_asynccnt)
    __builtin_amdgcn_s_wait_asynccnt(0);
#else
    asm volatile("s_wait_asynccnt 0x0" ::: "memory");
#endif
#endif
}

__device__ __forceinline__ v16h load_bfrag(const _Float16* cur, int lane, int nb) {
    return *(const v16h*)(cur + (nb * 32 + lane) * 16);
}

// ---------------------------------------------------------------------------
// Kernel 2: per-batch GEMM  A[b] = scores[b] (4096x4096, f32->f16) @ V[b]
// (4096x256, f16), f32 accumulation via v_wmma_f32_16x16x32_f16.
// Grid: (32 M-tiles, 8 batches), 256 threads = 8 waves.
// scores loads are non-temporal (streamed once; keep Vw/Aout L2-resident).
// Double-buffered LDS B tiles streamed via async global->LDS; B fragments
// register-pipelined 2 deep so ds latency hides under two WMMAs.
// ---------------------------------------------------------------------------
__global__ __launch_bounds__(256) void gemm_kernel(
    const float* __restrict__ S,        // scores [B][P][P]
    const _Float16* __restrict__ Vw,    // swizzled V [B][KT][BTILE_ELEMS]
    float* __restrict__ Aout) {         // [B][P][NN] f32

    __shared__ __align__(32) _Float16 Bsh[2][BTILE_ELEMS];  // 2 x 16 KB

    const int b     = blockIdx.y;
    const int mtile = blockIdx.x;
    const int tid   = threadIdx.x;
    const int wave  = tid >> 5;
    const int lane  = tid & 31;

    const int arow   = mtile * 128 + wave * 16 + (lane & 15);  // A row (M = lane&15)
    const int k_half = (lane >> 4) * 16;                       // A K sub-offset
    const float*    Srow = S  + ((long)b * PP + arow) * PP;
    const _Float16* Vb   = Vw + (long)b * KT * BTILE_ELEMS;

    v8f acc[16];
#pragma unroll
    for (int nb = 0; nb < 16; ++nb) acc[nb] = (v8f){0, 0, 0, 0, 0, 0, 0, 0};

    // ---- prologue: stage tile 0 into buffer 0 ----
    stage_btile((const uint4*)Vb, (uint4*)Bsh[0], tid);
    wait_stage();
    __syncthreads();

    for (int kt = 0; kt < KT; ++kt) {
        const _Float16* cur = Bsh[kt & 1];

        // ---- stage next tile into the other buffer (overlaps compute) ----
        if (kt + 1 < KT) {
            stage_btile((const uint4*)(Vb + (long)(kt + 1) * BTILE_ELEMS),
                        (uint4*)Bsh[(kt + 1) & 1], tid);
        }

        // ---- A fragment straight from global, non-temporal (f32 -> f16) ----
        // 16x32 f16 A layout used here: lanes 0-15 hold K {0..15},
        //                     lanes 16-31 hold K {16..31}; M = lane&15.
        const v4f* sp4 = (const v4f*)(Srow + (long)kt * KSTEP + k_half);
        v4f s0 = __builtin_nontemporal_load(sp4 + 0);   // K: k_half + 0..3
        v4f s1 = __builtin_nontemporal_load(sp4 + 1);   // K: k_half + 4..7
        v4f s2 = __builtin_nontemporal_load(sp4 + 2);   // K: k_half + 8..11
        v4f s3 = __builtin_nontemporal_load(sp4 + 3);   // K: k_half + 12..15
        if (kt + 1 < KT) {           // prefetch next scores tile (global_prefetch_b8)
            __builtin_prefetch(Srow + (long)(kt + 1) * KSTEP + k_half, 0, 0);
        }
        v16h a;
        a[0] = (_Float16)s0.x;  a[1] = (_Float16)s0.y;
        a[2] = (_Float16)s0.z;  a[3] = (_Float16)s0.w;
        a[4] = (_Float16)s1.x;  a[5] = (_Float16)s1.y;
        a[6] = (_Float16)s1.z;  a[7] = (_Float16)s1.w;
        a[8]  = (_Float16)s2.x; a[9]  = (_Float16)s2.y;
        a[10] = (_Float16)s2.z; a[11] = (_Float16)s2.w;
        a[12] = (_Float16)s3.x; a[13] = (_Float16)s3.y;
        a[14] = (_Float16)s3.z; a[15] = (_Float16)s3.w;

        // ---- 16 WMMAs, B fragments register-pipelined two ahead ----
        v16h f0 = load_bfrag(cur, lane, 0);
        v16h f1 = load_bfrag(cur, lane, 1);
#pragma unroll
        for (int nb = 0; nb < 16; ++nb) {
            v16h f2 = load_bfrag(cur, lane, (nb + 2) & 15);
            acc[nb] = __builtin_amdgcn_wmma_f32_16x16x32_f16(
                false, a, false, f0, (short)0, acc[nb], false, false);
            f0 = f1;
            f1 = f2;
        }

        wait_stage();     // this wave's async LDS writes for kt+1 landed
        __syncthreads();  // all waves done reading cur / writing next
    }

    // ---- epilogue: D layout  M = (lane>>4)*8 + r, N = nb*16 + (lane&15) ----
    const int mrow0 = mtile * 128 + wave * 16 + (lane >> 4) * 8;
    const int ncol  = lane & 15;
    float* Ab = Aout + (long)b * PP * NN;
#pragma unroll
    for (int nb = 0; nb < 16; ++nb) {
#pragma unroll
        for (int r = 0; r < 8; ++r) {
            Ab[(long)(mrow0 + r) * NN + nb * 16 + ncol] = acc[nb][r];
        }
    }
}

// ---------------------------------------------------------------------------
// Kernel 3: overlap-add fold of A back onto the HxW grid:
// out[b,c,y,x] = x + (alpha/4) * sum_{ky,kx valid} A[b, q(i,j), c*16+ky*4+kx]
// ---------------------------------------------------------------------------
__global__ void fold_add_kernel(const float* __restrict__ x,
                                const float* __restrict__ Aout,
                                const float* __restrict__ alpha,
                                float* __restrict__ out) {
    long gid = (long)blockIdx.x * blockDim.x + threadIdx.x;  // 2,097,152 total
    int xx = (int)(gid & 127);
    int yy = (int)((gid >> 7) & 127);
    int c  = (int)((gid >> 14) & 15);
    int b  = (int)(gid >> 18);

    const float* Ab = Aout + (long)b * PP * NN;
    const int ky0 = (yy + 1) & 1;
    const int kx0 = (xx + 1) & 1;

    float s = 0.f;
#pragma unroll
    for (int dy = 0; dy < 2; ++dy) {
        int ky = ky0 + 2 * dy;
        int i  = (yy + 1 - ky) >> 1;
        if (i < 0 || i > HS - 1) continue;
#pragma unroll
        for (int dx = 0; dx < 2; ++dx) {
            int kx = kx0 + 2 * dx;
            int j  = (xx + 1 - kx) >> 1;
            if (j < 0 || j > HS - 1) continue;
            int q = i * HS + j;
            int n = c * 16 + ky * 4 + kx;
            s += Ab[(long)q * NN + n];
        }
    }
    out[gid] = x[gid] + s * (alpha[0] * 0.25f);
}

// ---------------------------------------------------------------------------
extern "C" void kernel_launch(void* const* d_in, const int* in_sizes, int n_in,
                              void* d_out, int out_size, void* d_ws, size_t ws_size,
                              hipStream_t stream) {
    const float* x      = (const float*)d_in[0];  // [8,16,128,128]
    const float* scores = (const float*)d_in[1];  // [8,4096,4096]
    const float* alpha  = (const float*)d_in[2];  // [1]
    float*       out    = (float*)d_out;          // [8,16,128,128]

    _Float16* Vw   = (_Float16*)d_ws;                                  // 16 MB
    float*    Aout = (float*)((char*)d_ws + (size_t)16 * 1024 * 1024); // 32 MB

    // Kernel 1: 8*128*16*32*16 = 8,388,608 elements
    {
        long total = (long)BB * KT * BTILE_ELEMS;
        int blocks = (int)(total / 256);
        build_patches_kernel<<<blocks, 256, 0, stream>>>(x, Vw);
    }
    // Kernel 2: 32 M-tiles x 8 batches
    {
        dim3 grid(PP / 128, BB);
        gemm_kernel<<<grid, 256, 0, stream>>>(scores, Vw, Aout);
    }
    // Kernel 3: 8*16*128*128 = 2,097,152 outputs
    {
        long total = (long)BB * CC * HH * WW;
        int blocks = (int)(total / 256);
        fold_add_kernel<<<blocks, 256, 0, stream>>>(x, Aout, alpha, out);
    }
}
